// GroupedQueryAttention_28673201668554
// MI455X (gfx1250) — matrix operand order repaired and checked by reading, not compiled, not run on hardware
//
#include <hip/hip_runtime.h>
#include <hip/hip_bf16.h>

typedef __attribute__((ext_vector_type(16))) _Float16     v16h;
typedef __attribute__((ext_vector_type(8)))  _Float16     v8h;
typedef __attribute__((ext_vector_type(8)))  float        v8f;
typedef __attribute__((ext_vector_type(4)))  unsigned int u32x4;
typedef __attribute__((ext_vector_type(8)))  int          i32x8;
typedef __attribute__((ext_vector_type(4)))  int          i32x4;

#define SEQ      2048
#define DIM      4096
#define NH       32
#define NKV      8
#define HD       128
#define KVDIM    (NKV * HD)   // 1024

#if defined(__has_builtin)
#if __has_builtin(__builtin_amdgcn_tensor_load_to_lds)
#define TDM_OK 1
#endif
#endif
#ifndef TDM_OK
#define TDM_OK 0
#endif

// ---------------------------------------------------------------------------
// WMMA helper: D = A(16x32 f16) * B(32x16 f16) + C(16x16 f32)
// ---------------------------------------------------------------------------
__device__ __forceinline__ v8f wmma_f16(v16h a, v16h b, v8f c) {
  return __builtin_amdgcn_wmma_f32_16x16x32_f16(false, a, false, b, (short)0, c,
                                                false, false);
}

// A-fragment K-offset map (contiguous order, matches kkB): element j of v16h
__device__ __forceinline__ int kkA(int j, int khalf) {
  return (khalf << 4) + j;
}
// B-fragment K-offset map (contiguous order, matches kkA)
__device__ __forceinline__ int kkB(int j, int khalf) {
  return (khalf << 4) + j;
}

// ---------------------------------------------------------------------------
// TDM: DMA a (tile_rows x 128) f16 tile from global to LDS.
// LDS padding via D#: pad 4 DWORDs (16 B = 8 halves) every 64 DWORDs
// (256 B = one 128-elem f16 row)  ->  LDS row stride = HD + 8 halves.
// Descriptor layout per CDNA5 ISA ch.8 (groups 0/1; 2-D tensor, groups 2/3=0).
// ---------------------------------------------------------------------------
#if TDM_OK
__device__ __forceinline__ void tdm_load_rows_f16(unsigned lds_addr,
                                                  const _Float16* gptr,
                                                  unsigned tile_rows,
                                                  unsigned row_stride_elems) {
  unsigned long long ga = (unsigned long long)(size_t)gptr;
  u32x4 g0;
  g0[0] = 1u;                                   // count=1, user descriptor
  g0[1] = lds_addr;                             // LDS byte address
  g0[2] = (unsigned)(ga & 0xFFFFFFFFu);         // global_addr[31:0]
  g0[3] = (unsigned)((ga >> 32) & 0x01FFFFFFu)  // global_addr[56:32]
          | (2u << 30);                         // type = 2 ("image")
  i32x8 g1;
  g1[0] = (1 << 16)     // data_size = 2 bytes
        | (1 << 20)     // pad_enable
        | (5 << 22)     // pad_interval: every 64 DWORDs (one f16 row)
        | (3 << 25);    // pad_amount: 4 DWORDs (16 B)
  g1[1] = (int)((HD & 0xFFFFu) << 16);                    // tensor_dim0 lo16
  g1[2] = (int)((tile_rows & 0xFFFFu) << 16);             // dim0 hi / dim1 lo
  g1[3] = (int)((HD & 0xFFFFu) << 16);                    // dim1 hi / tile_dim0
  g1[4] = (int)(tile_rows & 0xFFFFu);                     // tile_dim1 (tile_dim2=0)
  g1[5] = (int)row_stride_elems;                          // tensor_dim0_stride lo32
  g1[6] = 0;                                              // stride hi / dim1_stride lo
  g1[7] = 0;
  i32x4 gz = {0, 0, 0, 0};
#if __clang_major__ >= 23
  i32x8 gz8 = {0, 0, 0, 0, 0, 0, 0, 0};
  __builtin_amdgcn_tensor_load_to_lds(g0, g1, gz, gz, gz8, 0);
#else
  __builtin_amdgcn_tensor_load_to_lds(g0, g1, gz, gz, 0);
#endif
}
#endif

// ---------------------------------------------------------------------------
// Tiled WMMA GEMM: C[MxN] = A[MxK] (f32) * B[KxN] (f32), f16 staged in LDS.
// Block: 128 threads (4 waves), tile 64x64, K-step 32. Dims are multiples.
// Global staging uses float4 (b128) loads; LDS stores merge to b64.
// ---------------------------------------------------------------------------
#define TM 64
#define TN 64
#define TK 32
#define ALD (TK + 8)
#define BLD (TN + 8)

template <bool OUT_F16>
__global__ __launch_bounds__(128) void gemm_kernel(
    const float* __restrict__ A, const float* __restrict__ B,
    _Float16* __restrict__ Cf16, float* __restrict__ Cf32,
    int M, int N, int K) {
  __shared__ _Float16 As[TM * ALD];
  __shared__ _Float16 Bs[TK * BLD];

  const int tid   = threadIdx.x;
  const int lane  = tid & 31;
  const int wave  = tid >> 5;
  const int lrow  = lane & 15;
  const int khalf = lane >> 4;
  const int row0  = blockIdx.y * TM;
  const int col0  = blockIdx.x * TN;

  v8f c[4];
#pragma unroll
  for (int nt = 0; nt < 4; ++nt)
#pragma unroll
    for (int r = 0; r < 8; ++r) c[nt][r] = 0.0f;

  for (int k0 = 0; k0 < K; k0 += TK) {
    // stage A tile (f32 -> f16), float4 global loads
#pragma unroll
    for (int i = 0; i < (TM * TK) / (128 * 4); ++i) {
      int c4 = tid + i * 128;              // float4 chunk id
      int r  = c4 / (TK / 4);
      int cc = (c4 % (TK / 4)) * 4;
      float4 v = *reinterpret_cast<const float4*>(
          &A[(size_t)(row0 + r) * K + k0 + cc]);
      As[r * ALD + cc + 0] = (_Float16)v.x;
      As[r * ALD + cc + 1] = (_Float16)v.y;
      As[r * ALD + cc + 2] = (_Float16)v.z;
      As[r * ALD + cc + 3] = (_Float16)v.w;
    }
    // stage B tile
#pragma unroll
    for (int i = 0; i < (TK * TN) / (128 * 4); ++i) {
      int c4 = tid + i * 128;
      int r  = c4 / (TN / 4);
      int cc = (c4 % (TN / 4)) * 4;
      float4 v = *reinterpret_cast<const float4*>(
          &B[(size_t)(k0 + r) * N + col0 + cc]);
      Bs[r * BLD + cc + 0] = (_Float16)v.x;
      Bs[r * BLD + cc + 1] = (_Float16)v.y;
      Bs[r * BLD + cc + 2] = (_Float16)v.z;
      Bs[r * BLD + cc + 3] = (_Float16)v.w;
    }
    __syncthreads();

    v16h a;
#pragma unroll
    for (int j = 0; j < 16; ++j)
      a[j] = As[(wave * 16 + lrow) * ALD + kkA(j, khalf)];

#pragma unroll
    for (int nt = 0; nt < 4; ++nt) {
      v16h b;
#pragma unroll
      for (int j = 0; j < 16; ++j)
        b[j] = Bs[kkB(j, khalf) * BLD + nt * 16 + lrow];
      c[nt] = wmma_f16(a, b, c[nt]);
    }
    __syncthreads();
  }

#pragma unroll
  for (int nt = 0; nt < 4; ++nt)
#pragma unroll
    for (int r = 0; r < 8; ++r) {
      int grow = row0 + wave * 16 + r + khalf * 8;
      int gcol = col0 + nt * 16 + lrow;
      if (OUT_F16)
        Cf16[(size_t)grow * N + gcol] = (_Float16)c[nt][r];
      else
        Cf32[(size_t)grow * N + gcol] = c[nt][r];
    }
}

// ---------------------------------------------------------------------------
// RoPE (in place, f16). Layout (s, h, d). cos/sin: (s, 128) f32.
// out[d]      = q[d]*cos[d]       - q[d+64]*sin[d]
// out[d+64]   = q[d+64]*cos[d+64] + q[d]   *sin[d+64]
// ---------------------------------------------------------------------------
__global__ void rope_kernel(_Float16* __restrict__ qk,
                            const float* __restrict__ cosb,
                            const float* __restrict__ sinb, int n_heads) {
  int idx = blockIdx.x * blockDim.x + threadIdx.x;
  int total = SEQ * n_heads * (HD / 2);
  if (idx >= total) return;
  int d = idx % (HD / 2);
  int h = (idx / (HD / 2)) % n_heads;
  int s = idx / ((HD / 2) * n_heads);
  size_t base = (size_t)s * n_heads * HD + (size_t)h * HD;
  float x1 = (float)qk[base + d];
  float x2 = (float)qk[base + d + HD / 2];
  float c1 = cosb[s * HD + d],          s1 = sinb[s * HD + d];
  float c2 = cosb[s * HD + d + HD / 2], s2 = sinb[s * HD + d + HD / 2];
  qk[base + d]          = (_Float16)(x1 * c1 - x2 * s1);
  qk[base + d + HD / 2] = (_Float16)(x2 * c2 + x1 * s2);
}

// ---------------------------------------------------------------------------
// Flash attention (causal, GQA 4:1). Block = 64 q-rows x 1 head, 128 threads
// (4 waves x 16 rows). K/Q staged by TDM DMA; V staged transposed (d-major)
// so P*V B-fragments are contiguous in LDS (vectorize to ds_load_b128).
// ---------------------------------------------------------------------------
#define BQ  64
#define BKV 32
#define QLD (HD + 8)
#define KLD (HD + 8)
#define VLD (BKV + 8)
#define PLD (BKV + 8)

__global__ __launch_bounds__(128) void attn_kernel(
    const _Float16* __restrict__ Q, const _Float16* __restrict__ Kh,
    const _Float16* __restrict__ Vh, float* __restrict__ Out) {
  __shared__ _Float16 Qs[BQ * QLD];
  __shared__ _Float16 Ks[BKV * KLD];
  __shared__ _Float16 Vt[HD * VLD];   // transposed: Vt[d][kv]
  __shared__ _Float16 Ps[4][16 * PLD];

  const int tid   = threadIdx.x;
  const int lane  = tid & 31;
  const int wave  = tid >> 5;
  const int lrow  = lane & 15;
  const int khalf = lane >> 4;
  const int h     = blockIdx.y;
  const int kvh   = h >> 2;                 // n_rep = 32/8 = 4
  const int q0    = blockIdx.x * BQ;

#if TDM_OK
  // SGPR wave index for TDM descriptor math (TDM reads D# from SGPRs)
  const int sw = __builtin_amdgcn_readfirstlane((int)threadIdx.x) >> 5;
  const unsigned ldsQ = (unsigned)(size_t)(void*)Qs;
  const unsigned ldsK = (unsigned)(size_t)(void*)Ks;

  // stage Q tile (64 x 128): each wave DMAs 16 rows
  tdm_load_rows_f16(ldsQ + (unsigned)(sw * 16 * QLD * 2),
                    Q + (size_t)(q0 + sw * 16) * DIM + (size_t)h * HD,
                    16u, (unsigned)DIM);
  __builtin_amdgcn_s_wait_tensorcnt(0);
#else
  for (int i = tid; i < BQ * HD; i += 128) {
    int r = i / HD, d = i % HD;
    Qs[r * QLD + d] = Q[(size_t)(q0 + r) * DIM + h * HD + d];
  }
#endif
  __syncthreads();

  // Q fragments for this wave's 16 rows (4 K-steps of 32)
  v16h qf[4];
#pragma unroll
  for (int ks = 0; ks < 4; ++ks)
#pragma unroll
    for (int j = 0; j < 16; ++j)
      qf[ks][j] = Qs[(wave * 16 + lrow) * QLD + ks * 32 + kkA(j, khalf)];

  v8f o[8];
#pragma unroll
  for (int n = 0; n < 8; ++n)
#pragma unroll
    for (int r = 0; r < 8; ++r) o[n][r] = 0.0f;

  float mrun[8], lrun[8];
#pragma unroll
  for (int r = 0; r < 8; ++r) { mrun[r] = -1e30f; lrun[r] = 0.0f; }

  const float scale = 0.08838834764831845f;  // 1/sqrt(128)
  const int ntiles = q0 / BKV + 2;           // causal: kv0 <= q0+63

  for (int t = 0; t < ntiles; ++t) {
    const int kv0 = t * BKV;
    __syncthreads();  // previous tile's K/V reads done before restage
#if TDM_OK
    // each wave DMAs 8 K rows, then waits its own TENSORcnt
    tdm_load_rows_f16(ldsK + (unsigned)(sw * 8 * KLD * 2),
                      Kh + (size_t)(kv0 + sw * 8) * KVDIM + (size_t)kvh * HD,
                      8u, (unsigned)KVDIM);
#else
    for (int i = tid; i < BKV * HD; i += 128) {
      int r = i / HD, d = i % HD;
      Ks[r * KLD + d] = Kh[(size_t)(kv0 + r) * KVDIM + kvh * HD + d];
    }
#endif
    // V staged transposed: b128 global reads, scalar transposed LDS scatter
#pragma unroll
    for (int i = 0; i < (BKV * HD) / (128 * 8); ++i) {   // 4 chunks/thread
      int c8  = tid + i * 128;                           // v8h chunk id
      int kv  = c8 / (HD / 8);
      int db  = (c8 % (HD / 8)) * 8;
      v8h v = *reinterpret_cast<const v8h*>(
          &Vh[(size_t)(kv0 + kv) * KVDIM + (size_t)kvh * HD + db]);
#pragma unroll
      for (int e = 0; e < 8; ++e) Vt[(db + e) * VLD + kv] = v[e];
    }
#if TDM_OK
    __builtin_amdgcn_s_wait_tensorcnt(0);
#endif
    __syncthreads();

    // S = Q * K^T : two 16x16 tiles, accumulate over HD=128 (4 K-steps)
    v8f s0, s1;
#pragma unroll
    for (int r = 0; r < 8; ++r) { s0[r] = 0.0f; s1[r] = 0.0f; }
#pragma unroll
    for (int ks = 0; ks < 4; ++ks) {
      v16h b0, b1;
#pragma unroll
      for (int j = 0; j < 16; ++j) {
        int kk = ks * 32 + kkB(j, khalf);
        b0[j] = Ks[lrow * KLD + kk];         // B[k][n] = K[n][k], n = lrow
        b1[j] = Ks[(16 + lrow) * KLD + kk];  // n = 16 + lrow
      }
      s0 = wmma_f16(qf[ks], b0, s0);
      s1 = wmma_f16(qf[ks], b1, s1);
    }

    // online softmax per row (rows live in 16-lane halves; xor<=8 stays in half)
    float p0[8], p1[8], alpha[8];
#pragma unroll
    for (int r = 0; r < 8; ++r) {
      int qrow = q0 + wave * 16 + r + khalf * 8;
      int c0 = kv0 + lrow, c1 = kv0 + 16 + lrow;
      float v0 = s0[r] * scale + (c0 > qrow ? -1e9f : 0.0f);
      float v1 = s1[r] * scale + (c1 > qrow ? -1e9f : 0.0f);
      float mx = fmaxf(v0, v1);
      mx = fmaxf(mx, __shfl_xor(mx, 1));
      mx = fmaxf(mx, __shfl_xor(mx, 2));
      mx = fmaxf(mx, __shfl_xor(mx, 4));
      mx = fmaxf(mx, __shfl_xor(mx, 8));
      float mnew = fmaxf(mrun[r], mx);
      float a = __expf(mrun[r] - mnew);
      float e0 = __expf(v0 - mnew);
      float e1 = __expf(v1 - mnew);
      float rs = e0 + e1;
      rs += __shfl_xor(rs, 1);
      rs += __shfl_xor(rs, 2);
      rs += __shfl_xor(rs, 4);
      rs += __shfl_xor(rs, 8);
      lrun[r] = lrun[r] * a + rs;
      mrun[r] = mnew;
      alpha[r] = a;
      p0[r] = e0;
      p1[r] = e1;
    }

    // P: C-layout -> per-wave LDS -> A-fragment (16x32)
    _Float16* myP = Ps[wave];
#pragma unroll
    for (int r = 0; r < 8; ++r) {
      myP[(r + khalf * 8) * PLD + lrow]      = (_Float16)p0[r];
      myP[(r + khalf * 8) * PLD + 16 + lrow] = (_Float16)p1[r];
    }
    v16h pf;
#pragma unroll
    for (int j = 0; j < 16; ++j)
      pf[j] = myP[lrow * PLD + kkA(j, khalf)];

    // O = O*alpha + P*V (8 N-tiles of 16 over HD, single K=32 step)
    // B[k][n] = V[kv=k][d=n] = Vt[(n)*VLD + k]  -> contiguous in k
#pragma unroll
    for (int n = 0; n < 8; ++n) {
#pragma unroll
      for (int r = 0; r < 8; ++r) o[n][r] *= alpha[r];
      v16h b;
#pragma unroll
      for (int j = 0; j < 16; ++j)
        b[j] = Vt[(n * 16 + lrow) * VLD + kkB(j, khalf)];
      o[n] = wmma_f16(pf, b, o[n]);
    }
  }

  // epilogue: divide by row sum, write (s, h*128+d) in f32
#pragma unroll
  for (int n = 0; n < 8; ++n)
#pragma unroll
    for (int r = 0; r < 8; ++r) {
      int qrow = q0 + wave * 16 + r + khalf * 8;
      Out[(size_t)qrow * DIM + h * HD + n * 16 + lrow] = o[n][r] / lrun[r];
    }
}

// ---------------------------------------------------------------------------
// Launcher
// ---------------------------------------------------------------------------
extern "C" void kernel_launch(void* const* d_in, const int* in_sizes, int n_in,
                              void* d_out, int out_size, void* d_ws,
                              size_t ws_size, hipStream_t stream) {
  const float* x    = (const float*)d_in[0];
  const float* cosb = (const float*)d_in[1];
  const float* sinb = (const float*)d_in[2];
  // d_in[3] = mask (computed inline: causal)
  const float* wq = (const float*)d_in[4];
  const float* wk = (const float*)d_in[5];
  const float* wv = (const float*)d_in[6];
  const float* wo = (const float*)d_in[7];
  float* out = (float*)d_out;

  char* ws = (char*)d_ws;
  _Float16* q_h = (_Float16*)ws;                                   // 16 MB
  _Float16* k_h = (_Float16*)(ws + (size_t)SEQ * DIM * 2);         //  4 MB
  _Float16* v_h = (_Float16*)(ws + (size_t)SEQ * DIM * 2 +
                              (size_t)SEQ * KVDIM * 2);            //  4 MB
  float* attn = (float*)(ws + (size_t)SEQ * DIM * 2 +
                         2 * (size_t)SEQ * KVDIM * 2);             // 32 MB

  // QKV projections (f32 in, f16 out, WMMA)
  gemm_kernel<true><<<dim3(DIM / TN, SEQ / TM), 128, 0, stream>>>(
      x, wq, q_h, nullptr, SEQ, DIM, DIM);
  gemm_kernel<true><<<dim3(KVDIM / TN, SEQ / TM), 128, 0, stream>>>(
      x, wk, k_h, nullptr, SEQ, KVDIM, DIM);
  gemm_kernel<true><<<dim3(KVDIM / TN, SEQ / TM), 128, 0, stream>>>(
      x, wv, v_h, nullptr, SEQ, KVDIM, DIM);

  // RoPE on Q and K
  int nq = SEQ * NH * (HD / 2);
  int nk = SEQ * NKV * (HD / 2);
  rope_kernel<<<(nq + 255) / 256, 256, 0, stream>>>(q_h, cosb, sinb, NH);
  rope_kernel<<<(nk + 255) / 256, 256, 0, stream>>>(k_h, cosb, sinb, NKV);

  // Flash attention (causal, GQA) — K/Q staged by Tensor Data Mover
  attn_kernel<<<dim3(SEQ / BQ, NH), 128, 0, stream>>>(q_h, k_h, v_h, attn);

  // Output projection (f32 out to d_out)
  gemm_kernel<false><<<dim3(DIM / TN, SEQ / TM), 128, 0, stream>>>(
      attn, wo, nullptr, out, SEQ, DIM, DIM);
}
